// NeuralMem_16157666968040
// MI455X (gfx1250) — matrix operand order repaired and checked by reading, not compiled, not run on hardware
//
#include <hip/hip_runtime.h>
#include <hip/hip_fp16.h>

typedef __attribute__((ext_vector_type(16))) _Float16 v16h;
typedef __attribute__((ext_vector_type(8)))  float    v8f;

#define K15    15
#define PADW   10
#define RSZ    128
#define L1d    134                 // 128 + 20 - 15 + 1
#define NPIX   (L1d * L1d)         // 17956 patch rows
#define MT     1123                // ceil(17956/16)
#define MPAD   (MT * 16)           // 17968
#define NMEM   16384
#define NT     (NMEM / 16)         // 1024 N-tiles
#define KDIM   225
#define KSTEPS 8                   // K padded to 256 = 8 * 32
#define NSPLIT 4                   // N-dimension split across grid.y
#define NTC    (NT / NSPLIT)       // 256 N-tiles per chunk
#define TILE_BYTES (KSTEPS * 32 * 32)   // 8192 B: one N-tile of packed B fragments
#define MTW    2                   // M-tiles per wave
#define MBLK   (8 * MTW)           // M-tiles per workgroup (8 waves)
#define MGRID  ((MT + MBLK - 1) / MBLK) // 71

// ---------------------------------------------------------------------------
// CDNA5 async global->LDS copy (ASYNCcnt-tracked), per-lane 16 bytes.
// ---------------------------------------------------------------------------
__device__ __forceinline__ void async_copy_b128(uint32_t lds_off, const void* gptr) {
    asm volatile("global_load_async_to_lds_b128 %0, %1, off"
                 :: "v"(lds_off), "v"((uint64_t)(uintptr_t)gptr)
                 : "memory");
}
__device__ __forceinline__ void wait_async(int n) {
    if (n >= 2)      asm volatile("s_wait_asynccnt 0x2" ::: "memory");
    else             asm volatile("s_wait_asynccnt 0x0" ::: "memory");
}
// LDS byte-offset of a __shared__ object (generic -> addrspace(3) -> int).
__device__ __forceinline__ uint32_t lds_off_of(void* p) {
    return (uint32_t)(uintptr_t)(__attribute__((address_space(3))) char*)(char*)p;
}

// padded+upsampled image value at pad-coords (y, x) in [0,148)
__device__ __forceinline__ float padval(const float* __restrict__ image, int y, int x) {
    if (y < PADW || y >= PADW + RSZ || x < PADW || x >= PADW + RSZ) return 0.0f;
    return image[((y - PADW) >> 1) * 64 + ((x - PADW) >> 1)];
}

// ---------------------------------------------------------------------------
// Pack X (17956 x 225 patch matrix) into WMMA A-fragments (16-bit 16x32 layout).
// ---------------------------------------------------------------------------
__global__ __launch_bounds__(256) void pack_a_kernel(const float* __restrict__ image,
                                                     _Float16* __restrict__ apack) {
    int t     = blockIdx.x * blockDim.x + threadIdx.x;   // 0 .. MT*8*32-1
    int lane  = t & 31;
    int kstep = (t >> 5) & 7;
    int mt    = t >> 8;
    int m     = mt * 16 + (lane & 15);
    int kgrp  = lane >> 4;

    bool mvalid = (m < NPIX);
    int r = 0, c = 0;
    if (mvalid) { r = m / L1d; c = m % L1d; }

    v16h frag;
    for (int h = 0; h < 16; ++h) {
        int v  = h >> 1;
        int tt = h & 1;
        int k  = (v < 4) ? (32 * kstep + 8 * kgrp + 2 * v + tt)
                         : (32 * kstep + 16 + 8 * kgrp + 2 * (v - 4) + tt);
        float x = 0.0f;
        if (mvalid && k < KDIM) {
            int i = k / K15, j = k % K15;
            x = padval(image, r + i, c + j);
        }
        frag[h] = (_Float16)x;
    }
    ((v16h*)apack)[t] = frag;
}

// ---------------------------------------------------------------------------
// Pack mem (16384 x 225) into WMMA B-fragments (16-bit 32x16 layout).
// ---------------------------------------------------------------------------
__global__ __launch_bounds__(256) void pack_b_kernel(const float* __restrict__ mem,
                                                     _Float16* __restrict__ bpack) {
    int t     = blockIdx.x * blockDim.x + threadIdx.x;   // 0 .. NT*8*32-1
    int lane  = t & 31;
    int kstep = (t >> 5) & 7;
    int nt    = t >> 8;
    int n     = nt * 16 + (lane & 15);
    int khalf = lane >> 4;

    v16h frag;
    for (int h = 0; h < 16; ++h) {
        int v  = h >> 1;
        int tt = h & 1;
        int k  = (v < 4) ? (32 * kstep + 8 * khalf + 2 * v + tt)
                         : (32 * kstep + 16 + 8 * khalf + 2 * (v - 4) + tt);
        float x = (k < KDIM) ? mem[n * KDIM + k] : 0.0f;
        frag[h] = (_Float16)x;
    }
    ((v16h*)bpack)[t] = frag;
}

// ---------------------------------------------------------------------------
// Squared norms of mem rows.
// ---------------------------------------------------------------------------
__global__ __launch_bounds__(256) void norms_kernel(const float* __restrict__ mem,
                                                    float* __restrict__ norms) {
    int n = blockIdx.x * blockDim.x + threadIdx.x;
    if (n >= NMEM) return;
    float s = 0.0f;
    const float* row = mem + n * KDIM;
    for (int k = 0; k < KDIM; ++k) s += row[k] * row[k];
    norms[n] = s;
}

// ---------------------------------------------------------------------------
// Fused GEMM + partial argmin over one N-chunk.
// 8 waves/block; each wave owns 2 M-tiles (A pinned in VGPRs). B tiles are
// staged in LDS by async-to-LDS DMA, TRIPLE-buffered (one barrier per tile),
// and shared by all 8 waves. B fragments are software-pipelined through
// registers so LDS latency overlaps the WMMAs.
// ---------------------------------------------------------------------------
__global__ __launch_bounds__(256) void gemm_argmin_kernel(const _Float16* __restrict__ apack,
                                                          const _Float16* __restrict__ bpack,
                                                          const float* __restrict__ norms,
                                                          float* __restrict__ pbd,
                                                          int* __restrict__ pbn) {
    __shared__ __align__(128) char bstage[3 * TILE_BYTES];

    const int tid   = threadIdx.x;
    const int lane  = tid & 31;
    const int wave  = tid >> 5;
    const int chunk = blockIdx.y;

    const int gw  = blockIdx.x * 8 + wave;        // global wave id
    int mt0 = gw * MTW;                           // clamped, no early exit:
    int mt1 = gw * MTW + 1;                       // barriers need all waves alive
    if (mt0 > MT - 1) mt0 = MT - 1;
    if (mt1 > MT - 1) mt1 = MT - 1;

    const v16h* A = (const v16h*)apack;

    v16h a0[KSTEPS], a1[KSTEPS];
#pragma unroll
    for (int ks = 0; ks < KSTEPS; ++ks) {
        a0[ks] = A[(mt0 * KSTEPS + ks) * 32 + lane];
        a1[ks] = A[(mt1 * KSTEPS + ks) * 32 + lane];
    }

    float bd0[8], bd1[8];
    int   bn0[8], bn1[8];
#pragma unroll
    for (int r = 0; r < 8; ++r) {
        bd0[r] = 3.4e38f; bd1[r] = 3.4e38f; bn0[r] = 0; bn1[r] = 0;
    }

    const int ncol = lane & 15;
    const uint32_t lds_base = lds_off_of(bstage);
    const uint32_t lane_off = (uint32_t)tid * 32u;
    const char* srcbase = (const char*)bpack + (size_t)(chunk * NTC) * TILE_BYTES
                                             + (size_t)tid * 32u;

    // Prologue: stage tiles 0 and 1 into buffers 0 and 1.
    async_copy_b128(lds_base + lane_off,      srcbase);
    async_copy_b128(lds_base + lane_off + 16, srcbase + 16);
    async_copy_b128(lds_base + TILE_BYTES + lane_off,      srcbase + TILE_BYTES);
    async_copy_b128(lds_base + TILE_BYTES + lane_off + 16, srcbase + TILE_BYTES + 16);

    int cur  = 0;   // buffer holding tile t
    int nxt2 = 2;   // buffer that will hold tile t+2
    for (int t = 0; t < NTC; ++t) {
        // Own DMAs for tile t complete (tile t+1 may still be in flight).
        wait_async((t + 1 < NTC) ? 2 : 0);
        __syncthreads();   // all waves' tile-t DMAs visible; all waves done with buf[nxt2]

        if (t + 2 < NTC) { // safe: every wave has finished consuming buf[nxt2] (= tile t-1)
            uint32_t d  = lds_base + (uint32_t)nxt2 * TILE_BYTES + lane_off;
            const char* s = srcbase + (size_t)(t + 2) * TILE_BYTES;
            async_copy_b128(d,      s);
            async_copy_b128(d + 16, s + 16);
        }

        const v16h* bf = (const v16h*)(bstage + cur * TILE_BYTES);
        const int   n  = (chunk * NTC + t) * 16 + ncol;
        const float nn = norms[n];

        v8f c0 = {}, c1 = {};
        v16h b = bf[lane];                      // k-step 0
#pragma unroll
        for (int ks = 0; ks < KSTEPS - 1; ++ks) {
            v16h bnx = bf[(ks + 1) * 32 + lane];  // prefetch next fragment
            c0 = __builtin_amdgcn_wmma_f32_16x16x32_f16(false, a0[ks], false, b,
                                                        (short)0, c0, false, false);
            c1 = __builtin_amdgcn_wmma_f32_16x16x32_f16(false, a1[ks], false, b,
                                                        (short)0, c1, false, false);
            b = bnx;
        }
        c0 = __builtin_amdgcn_wmma_f32_16x16x32_f16(false, a0[KSTEPS - 1], false, b,
                                                    (short)0, c0, false, false);
        c1 = __builtin_amdgcn_wmma_f32_16x16x32_f16(false, a1[KSTEPS - 1], false, b,
                                                    (short)0, c1, false, false);

#pragma unroll
        for (int r = 0; r < 8; ++r) {
            float d0 = __builtin_fmaf(-2.0f, c0[r], nn);
            float d1 = __builtin_fmaf(-2.0f, c1[r], nn);
            if (d0 < bd0[r]) { bd0[r] = d0; bn0[r] = n; }  // strict <: first-min wins
            if (d1 < bd1[r]) { bd1[r] = d1; bn1[r] = n; }
        }

        cur  = (cur  == 2) ? 0 : cur  + 1;
        nxt2 = (nxt2 == 2) ? 0 : nxt2 + 1;
    }

    // Argmin reduction across the 16 lanes sharing each row (xor butterfly).
#pragma unroll
    for (int mask = 1; mask <= 8; mask <<= 1) {
#pragma unroll
        for (int r = 0; r < 8; ++r) {
            float od0 = __shfl_xor(bd0[r], mask, 32);
            int   on0 = __shfl_xor(bn0[r], mask, 32);
            if (od0 < bd0[r] || (od0 == bd0[r] && on0 < bn0[r])) { bd0[r] = od0; bn0[r] = on0; }
            float od1 = __shfl_xor(bd1[r], mask, 32);
            int   on1 = __shfl_xor(bn1[r], mask, 32);
            if (od1 < bd1[r] || (od1 == bd1[r] && on1 < bn1[r])) { bd1[r] = od1; bn1[r] = on1; }
        }
    }

    if (ncol == 0) {
        const int half = lane >> 4;   // lane0 -> rows 0..7, lane16 -> rows 8..15
        const size_t base = (size_t)chunk * MPAD;
#pragma unroll
        for (int r = 0; r < 8; ++r) {
            int m0 = mt0 * 16 + 8 * half + r;
            if (m0 < NPIX) { pbd[base + m0] = bd0[r]; pbn[base + m0] = bn0[r]; }
            int m1 = mt1 * 16 + 8 * half + r;
            if (m1 < NPIX) { pbd[base + m1] = bd1[r]; pbn[base + m1] = bn1[r]; }
        }
    }
}

// ---------------------------------------------------------------------------
// Combine per-chunk partial argmins (chunks ordered by n => tie -> smallest n).
// ---------------------------------------------------------------------------
__global__ __launch_bounds__(256) void combine_kernel(const float* __restrict__ pbd,
                                                      const int* __restrict__ pbn,
                                                      int* __restrict__ idx) {
    int m = blockIdx.x * blockDim.x + threadIdx.x;
    if (m >= NPIX) return;
    float bd = pbd[m];
    int   bn = pbn[m];
    for (int c = 1; c < NSPLIT; ++c) {
        float d = pbd[(size_t)c * MPAD + m];
        int   n = pbn[(size_t)c * MPAD + m];
        if (d < bd || (d == bd && n < bn)) { bd = d; bn = n; }
    }
    idx[m] = bn;
}

// ---------------------------------------------------------------------------
// Overlap-add fold, directly producing the strided 64x64 output samples.
// ---------------------------------------------------------------------------
__global__ __launch_bounds__(256) void fold_kernel(const float* __restrict__ mem,
                                                   const int* __restrict__ idx,
                                                   float* __restrict__ raw) {
    int t = blockIdx.x * blockDim.x + threadIdx.x;   // 0..4095
    int a = t >> 6, b = t & 63;
    int y = PADW + 2 * a, x = PADW + 2 * b;
    float acc = 0.0f;
    for (int i = 0; i < K15; ++i) {
        int r = y - i;
        if (r < 0 || r >= L1d) continue;
        for (int j = 0; j < K15; ++j) {
            int c = x - j;
            if (c < 0 || c >= L1d) continue;
            int pix = r * L1d + c;
            acc += mem[(size_t)idx[pix] * KDIM + i * K15 + j];
        }
    }
    raw[t] = acc;
}

// ---------------------------------------------------------------------------
// Single-block max reduction + normalize.
// ---------------------------------------------------------------------------
__global__ __launch_bounds__(256) void normalize_kernel(const float* __restrict__ raw,
                                                        float* __restrict__ out) {
    __shared__ float smax[256];
    float m = -3.4e38f;
    for (int i = threadIdx.x; i < 4096; i += 256) m = fmaxf(m, raw[i]);
    smax[threadIdx.x] = m;
    __syncthreads();
    for (int s = 128; s > 0; s >>= 1) {
        if ((int)threadIdx.x < s) smax[threadIdx.x] = fmaxf(smax[threadIdx.x], smax[threadIdx.x + s]);
        __syncthreads();
    }
    float mv = smax[0];
    for (int i = threadIdx.x; i < 4096; i += 256) out[i] = raw[i] / mv;
}

extern "C" void kernel_launch(void* const* d_in, const int* in_sizes, int n_in,
                              void* d_out, int out_size, void* d_ws, size_t ws_size,
                              hipStream_t stream) {
    const float* image = (const float*)d_in[0];   // 64*64 f32
    const float* mem   = (const float*)d_in[1];   // 16384*225 f32
    float* out = (float*)d_out;                   // 64*64 f32

    char* ws = (char*)d_ws;
    size_t off = 0;
    auto carve = [&](size_t bytes) { void* p = ws + off;
                                     off = (off + bytes + 1023) & ~(size_t)1023; return p; };
    _Float16* apack = (_Float16*)carve((size_t)MT * KSTEPS * 32 * 16 * sizeof(_Float16)); // 9.2 MB
    _Float16* bpack = (_Float16*)carve((size_t)NT * KSTEPS * 32 * 16 * sizeof(_Float16)); // 8.4 MB
    float*    norms = (float*)carve((size_t)NMEM * sizeof(float));
    float*    pbd   = (float*)carve((size_t)NSPLIT * MPAD * sizeof(float));
    int*      pbn   = (int*)carve((size_t)NSPLIT * MPAD * sizeof(int));
    int*      idx   = (int*)carve((size_t)MPAD * sizeof(int));
    float*    raw   = (float*)carve((size_t)4096 * sizeof(float));
    (void)off; (void)ws_size; (void)in_sizes; (void)n_in; (void)out_size;

    pack_a_kernel<<<MT, 256, 0, stream>>>(image, apack);
    pack_b_kernel<<<NT, 256, 0, stream>>>(mem, bpack);
    norms_kernel<<<(NMEM + 255) / 256, 256, 0, stream>>>(mem, norms);
    gemm_argmin_kernel<<<dim3(MGRID, NSPLIT), 256, 0, stream>>>(apack, bpack, norms, pbd, pbn);
    combine_kernel<<<(NPIX + 255) / 256, 256, 0, stream>>>(pbd, pbn, idx);
    fold_kernel<<<16, 256, 0, stream>>>(mem, idx, raw);
    normalize_kernel<<<1, 256, 0, stream>>>(raw, out);
}
